// RegionProposalNetwork_15899968930427
// MI455X (gfx1250) — matrix operand order repaired and checked by reading, not compiled, not run on hardware
//
#include <hip/hip_runtime.h>
#include <hip/hip_bf16.h>

#define NIMG   8
#define CIN    512
#define HH     64
#define WW     64
#define HWPIX  (HH*WW)        // 4096
#define NANC   9
#define KTOT   (CIN*9)        // 4608
#define ANCN   (HWPIX*NANC)   // 36864
#define PRE_NMS  2000
#define POST_NMS 300

typedef __attribute__((ext_vector_type(16))) __bf16 v16bf;
typedef __attribute__((ext_vector_type(8)))  float  v8f;

static __device__ __forceinline__ unsigned f2bf(float f) {
    unsigned u = __float_as_uint(f);
    unsigned r = u + 0x7FFFu + ((u >> 16) & 1u);   // round-to-nearest-even
    return r >> 16;
}
static __device__ __forceinline__ unsigned fkey(float f) {
    unsigned u = __float_as_uint(f);
    return (u & 0x80000000u) ? ~u : (u | 0x80000000u);
}

// ---------------------------------------------------------------------------
// Kernel 0a: weights -> bf16 pairs, tap-major K:  Wp[co][r][cipair]
// ---------------------------------------------------------------------------
__global__ void wpack_k(const float* __restrict__ w, unsigned* __restrict__ Wp)
{
    int id = blockIdx.x * blockDim.x + threadIdx.x;      // 512*9*256
    if (id >= CIN * 9 * 256) return;
    int cp = id & 255;
    int t  = id >> 8;
    int rr = t % 9;
    int co = t / 9;
    int ci = cp * 2;
    float f0 = w[((size_t)co * CIN + ci)     * 9 + rr];
    float f1 = w[((size_t)co * CIN + ci + 1) * 9 + rr];
    Wp[id] = f2bf(f0) | (f2bf(f1) << 16);
}

// ---------------------------------------------------------------------------
// Kernel 0b: activations NCHW f32 -> NHWC bf16 pairs:  Xp[pix][cipair]
// LDS transpose so both global read and write are coalesced.
// ---------------------------------------------------------------------------
__global__ __launch_bounds__(256)
void xpack_k(const float* __restrict__ x, unsigned* __restrict__ Xp)
{
    __shared__ float tile[64 * 64];                      // [ci_l][xcol]
    const int row = blockIdx.x;                          // (n, y)
    const int n = row >> 6, y = row & 63;
    for (int ci0 = 0; ci0 < CIN; ci0 += 64) {
        for (int i = threadIdx.x; i < 4096; i += 256) {
            int cl = i >> 6, xc = i & 63;
            tile[i] = x[(((size_t)n * CIN + ci0 + cl) * HH + y) * WW + xc];
        }
        __syncthreads();
        for (int i = threadIdx.x; i < 2048; i += 256) {
            int xc = i >> 5, cpl = i & 31;
            unsigned pk = f2bf(tile[(cpl * 2) * 64 + xc]) |
                          (f2bf(tile[(cpl * 2 + 1) * 64 + xc]) << 16);
            Xp[((size_t)n * HWPIX + y * WW + xc) * 256 + (ci0 >> 1) + cpl] = pk;
        }
        __syncthreads();
    }
}

// ---------------------------------------------------------------------------
// Kernel 1: 3x3 conv as implicit GEMM via v_wmma_f32_16x16x32_bf16
// K tap-major: per 32-wide k-step the tap is uniform (scalar im2col); OOB rows
// skip whole taps uniformly. WG tile 256co x 64px, wave tile 32co x 64px:
// 8 WMMAs per wave per k-step (2 A-frags x 4 B-frags). All fragment loads are
// explicit uint4 reads so they lower to ds_load_b128.
// Grid: (512 image-rows, 2 co-blocks of 256). Block: 256 threads = 8 waves.
// ---------------------------------------------------------------------------
__global__ __launch_bounds__(256)
void conv3x3_wmma(const unsigned* __restrict__ Wp,   // [co][9][256] bf16x2
                  const unsigned* __restrict__ Xp,   // [pix][256]   bf16x2
                  const float* __restrict__ bias,
                  float* __restrict__ hout)
{
    __shared__ alignas(16) unsigned As[256 * 16];    // [co][kpair]  16KB
    __shared__ alignas(16) unsigned Bs[64 * 16];     // [col][kpair]  4KB

    const int row = blockIdx.x;                      // (n, y)
    const int n   = row >> 6;
    const int y   = row & 63;
    const int cob = blockIdx.y * 256;

    const int tid  = threadIdx.x;
    const int wav  = tid >> 5;
    const int lane = tid & 31;
    const int m    = lane & 15;
    const int hi   = lane >> 4;

    union F { v16bf v; uint4 q[2]; };
    F ua0, ua1, ub;
    v8f c[8] = {};

    // per-thread B-load geometry (constant)
    const int bcol = tid >> 2;
    const int bkp4 = (tid & 3) * 4;

    for (int rr = 0; rr < 9; ++rr) {
        const int ky = rr / 3;
        const int dx = (rr - ky * 3) - 1;
        const int yy = y + ky - 1;
        if ((unsigned)yy >= 64u) continue;           // block-uniform: whole tap is zero
        const unsigned* wbase = Wp + ((size_t)cob * 9 + rr) * 256;
        const unsigned* xbase = Xp + ((size_t)n * HWPIX + (size_t)yy * WW) * 256;

        const int xx = bcol + dx;
        const bool bok = (unsigned)xx < 64u;
        const unsigned* bptr = xbase + (size_t)(bok ? xx : 0) * 256 + bkp4;

        for (int cp0 = 0; cp0 < 256; cp0 += 16) {    // 16 ci-pairs = K-step of 32
            // A tile: 256co x 16kp, b128 loads (4 per thread)
            #pragma unroll
            for (int e = 0; e < 4; ++e) {
                int idx = e * 256 + tid;             // 0..1023 quads
                int co  = idx >> 2;
                int kp4 = (idx & 3) * 4;
                uint4 v = *(const uint4*)(wbase + (size_t)co * (9 * 256) + cp0 + kp4);
                *(uint4*)&As[co * 16 + kp4] = v;
            }
            // B tile: 64col x 16kp, b128 load (1 per thread), edge-col predicate
            {
                uint4 v = make_uint4(0u, 0u, 0u, 0u);
                if (bok) v = *(const uint4*)(bptr + cp0);
                *(uint4*)&Bs[bcol * 16 + bkp4] = v;
            }
            // prefetch next k-step's B quad toward this WGP
            if (bok && cp0 + 16 < 256)
                __builtin_prefetch(bptr + cp0 + 16, 0, 3);
            __syncthreads();

            // A fragments (ISA 16-bit 16x32): lanes<16 K=0..7,16..23; else 8..15,24..31
            // all 16B-aligned -> ds_load_b128
            const int kpo = hi * 4;
            const int ar0 = (wav * 32 + m) * 16;
            const int ar1 = (wav * 32 + 16 + m) * 16;
            ua0.q[0] = *(const uint4*)&As[ar0 + kpo];
            ua0.q[1] = *(const uint4*)&As[ar0 + 8 + kpo];
            ua1.q[0] = *(const uint4*)&As[ar1 + kpo];
            ua1.q[1] = *(const uint4*)&As[ar1 + 8 + kpo];

            #pragma unroll
            for (int s = 0; s < 4; ++s) {
                const int brow = (s * 16 + m) * 16 + hi * 4;
                ub.q[0] = *(const uint4*)&Bs[brow];
                ub.q[1] = *(const uint4*)&Bs[brow + 8];
                c[s]     = __builtin_amdgcn_wmma_f32_16x16x32_bf16(
                               false, ua0.v, false, ub.v, (short)0, c[s],     false, false);
                c[4 + s] = __builtin_amdgcn_wmma_f32_16x16x32_bf16(
                               false, ua1.v, false, ub.v, (short)0, c[4 + s], false, false);
            }
            __syncthreads();
        }
    }

    // epilogue: bias + ReLU, f32 store to NCHW h
    #pragma unroll
    for (int cc = 0; cc < 2; ++cc) {
        #pragma unroll
        for (int s = 0; s < 4; ++s) {
            int col = s * 16 + m;
            #pragma unroll
            for (int v = 0; v < 8; ++v) {
                int co = cob + wav * 32 + cc * 16 + v + hi * 8;
                float r = c[cc * 4 + s][v] + bias[co];
                hout[(((size_t)n * CIN + co) * HH + y) * WW + col] = r > 0.f ? r : 0.f;
            }
        }
    }
}

// ---------------------------------------------------------------------------
// Kernel 2: 1x1 heads + softmax. One block per image-row, thread per column.
// ---------------------------------------------------------------------------
__global__ __launch_bounds__(64)
void conv1x1_head(const float* __restrict__ h,
                  const float* __restrict__ score_w, const float* __restrict__ score_b,
                  const float* __restrict__ loc_w,   const float* __restrict__ loc_b,
                  float* __restrict__ out_locs,
                  float* __restrict__ out_scores,
                  float* __restrict__ fg)
{
    __shared__ float hs[128 * 64];
    const int row = blockIdx.x;
    const int n   = row >> 6;
    const int y   = row & 63;
    const int x   = threadIdx.x;

    float acc[54];
    #pragma unroll
    for (int i = 0; i < 54; ++i) acc[i] = 0.f;

    for (int c0 = 0; c0 < CIN; c0 += 128) {
        for (int i = threadIdx.x; i < 128 * 64; i += 64)
            hs[i] = h[(((size_t)n * CIN + c0 + (i >> 6)) * HH + y) * WW + (i & 63)];
        __syncthreads();
        for (int cl = 0; cl < 128; ++cl) {
            float hv = hs[cl * 64 + x];
            int ci = c0 + cl;
            #pragma unroll
            for (int ch = 0; ch < 36; ++ch) acc[ch]      += hv * loc_w[ch * CIN + ci];
            #pragma unroll
            for (int ch = 0; ch < 18; ++ch) acc[36 + ch] += hv * score_w[ch * CIN + ci];
        }
        __syncthreads();
    }

    const int p = y * WW + x;
    for (int a = 0; a < 9; ++a) {
        size_t lb = ((size_t)n * ANCN + (size_t)p * 9 + a) * 4;
        #pragma unroll
        for (int d = 0; d < 4; ++d)
            out_locs[lb + d] = acc[a * 4 + d] + loc_b[a * 4 + d];

        float s0 = acc[36 + a * 2 + 0] + score_b[a * 2 + 0];
        float s1 = acc[36 + a * 2 + 1] + score_b[a * 2 + 1];
        size_t sb = ((size_t)n * ANCN + (size_t)p * 9 + a) * 2;
        out_scores[sb + 0] = s0;
        out_scores[sb + 1] = s1;
        float mx = fmaxf(s0, s1);
        float e0 = __expf(s0 - mx), e1 = __expf(s1 - mx);
        fg[(size_t)n * ANCN + (size_t)p * 9 + a] = e1 / (e0 + e1);
    }
}

// ---------------------------------------------------------------------------
// Kernel 3: anchors
// ---------------------------------------------------------------------------
__global__ void anchors_k(float* __restrict__ anc)
{
    int i = blockIdx.x * blockDim.x + threadIdx.x;
    if (i >= ANCN) return;
    int p = i / 9, a = i - p * 9;
    int ri = a / 3, si = a - ri * 3;
    const float ratios[3] = {0.5f, 1.f, 2.f};
    const float scales[3] = {8.f, 16.f, 32.f};
    float r = ratios[ri], s = scales[si];
    float hh2 = 16.f * s * sqrtf(r)       * 0.5f;
    float ww2 = 16.f * s * sqrtf(1.f / r) * 0.5f;
    float sy = (float)((p >> 6) * 16);
    float sx = (float)((p & 63) * 16);
    anc[i * 4 + 0] = sy + 8.f - hh2;
    anc[i * 4 + 1] = sx + 8.f - ww2;
    anc[i * 4 + 2] = sy + 8.f + hh2;
    anc[i * 4 + 3] = sx + 8.f + ww2;
}

// ---------------------------------------------------------------------------
// Kernel 4: loc2bbox + clip + min-size filter + monotonic sort key
// ---------------------------------------------------------------------------
__global__ void decode_k(const float* __restrict__ anc,
                         const float* __restrict__ locs,
                         const float* __restrict__ fg,
                         float* __restrict__ boxes,
                         unsigned* __restrict__ keys)
{
    int i = blockIdx.x * blockDim.x + threadIdx.x;
    if (i >= NIMG * ANCN) return;
    int j = i % ANCN;
    float a0 = anc[j * 4 + 0], a1 = anc[j * 4 + 1];
    float a2 = anc[j * 4 + 2], a3 = anc[j * 4 + 3];
    float ah = a2 - a0, aw = a3 - a1;
    float cy = a0 + 0.5f * ah, cx = a1 + 0.5f * aw;
    const float* l = locs + (size_t)i * 4;
    float cy2 = l[0] * ah + cy;
    float cx2 = l[1] * aw + cx;
    float h2  = __expf(l[2]) * ah;
    float w2  = __expf(l[3]) * aw;
    float b0 = fminf(fmaxf(cy2 - 0.5f * h2, 0.f), 1024.f);
    float b1 = fminf(fmaxf(cx2 - 0.5f * w2, 0.f), 1024.f);
    float b2 = fminf(fmaxf(cy2 + 0.5f * h2, 0.f), 1024.f);
    float b3 = fminf(fmaxf(cx2 + 0.5f * w2, 0.f), 1024.f);
    boxes[(size_t)i * 4 + 0] = b0;
    boxes[(size_t)i * 4 + 1] = b1;
    boxes[(size_t)i * 4 + 2] = b2;
    boxes[(size_t)i * 4 + 3] = b3;
    float f = fg[i];
    if (!((b2 - b0) >= 16.f && (b3 - b1) >= 16.f)) f = -INFINITY;
    keys[i] = fkey(f);
}

// ---------------------------------------------------------------------------
// Kernel 5: per-image top-2000 (radix-select) + bitonic sort + NMS + emit rois
// ---------------------------------------------------------------------------
__global__ __launch_bounds__(256)
void propose_k(const unsigned* __restrict__ keys,
               const float* __restrict__ boxes,
               float* __restrict__ rois,
               int* __restrict__ roi_idx)
{
    __shared__ unsigned hist[256];
    __shared__ unsigned s_prefix, s_need, s_T, s_tq, s_cg, cntHigh, cntTie;
    __shared__ unsigned selKey[2048];
    __shared__ int      selIdx[2048];
    __shared__ float    bb[PRE_NMS * 4];
    __shared__ unsigned char keep[2048];

    const int img = blockIdx.x;
    const int tid = threadIdx.x;
    const unsigned* k = keys + (size_t)img * ANCN;
    const float* bsrc = boxes + (size_t)img * ANCN * 4;

    if (tid == 0) { s_prefix = 0u; s_need = PRE_NMS; }
    __syncthreads();

    // exact top-2000 threshold: 4 passes of 8-bit radix select (MSB first)
    for (int pass = 0; pass < 4; ++pass) {
        hist[tid] = 0u;
        __syncthreads();
        const int shift = 24 - 8 * pass;
        const unsigned pref = s_prefix;
        for (int j = tid; j < ANCN; j += 256) {
            unsigned kv = k[j];
            bool ok = (pass == 0) || ((kv >> (shift + 8)) == pref);
            if (ok) atomicAdd(&hist[(kv >> shift) & 255u], 1u);
        }
        __syncthreads();
        if (tid == 0) {
            unsigned need = s_need, cum = 0; int b = 255;
            for (; b > 0; --b) {
                unsigned c = hist[b];
                if (cum + c >= need) break;
                cum += c;
            }
            s_prefix = (pref << 8) | (unsigned)b;
            s_need   = need - cum;
        }
        __syncthreads();
    }
    if (tid == 0) {
        s_T = s_prefix; s_tq = s_need; s_cg = PRE_NMS - s_need;
        cntHigh = 0u; cntTie = 0u;
    }
    __syncthreads();
    const unsigned T = s_T, tq = s_tq, cg = s_cg;

    // compaction: strict-greater first, then quota of ties
    for (int j = tid; j < ANCN; j += 256) {
        unsigned kv = k[j];
        if (kv > T) {
            unsigned pos = atomicAdd(&cntHigh, 1u);
            selKey[pos] = kv; selIdx[pos] = j;
        } else if (kv == T) {
            unsigned pos = atomicAdd(&cntTie, 1u);
            if (pos < tq) { selKey[cg + pos] = kv; selIdx[cg + pos] = j; }
        }
    }
    __syncthreads();
    for (int i = PRE_NMS + tid; i < 2048; i += 256) { selKey[i] = 0u; selIdx[i] = -1; }
    __syncthreads();

    // bitonic sort: key descending, idx ascending on ties (deterministic)
    for (unsigned k2 = 2; k2 <= 2048; k2 <<= 1) {
        for (unsigned jj = k2 >> 1; jj > 0; jj >>= 1) {
            for (unsigned i = tid; i < 2048; i += 256) {
                unsigned x2 = i ^ jj;
                if (x2 > i) {
                    unsigned ka = selKey[i], kb = selKey[x2];
                    int ia = selIdx[i], ib = selIdx[x2];
                    bool aFirst = (ka > kb) || (ka == kb && ia < ib);
                    bool up = ((i & k2) == 0);
                    if (up ? !aFirst : aFirst) {
                        selKey[i] = kb; selKey[x2] = ka;
                        selIdx[i] = ib; selIdx[x2] = ia;
                    }
                }
            }
            __syncthreads();
        }
    }

    // gather boxes + keep mask (keep iff score > -inf)
    const unsigned NEGKEY = 0x007FFFFFu;   // fkey(-inf)
    for (int i = tid; i < 2048; i += 256) {
        keep[i] = (i < PRE_NMS && selKey[i] > NEGKEY) ? 1 : 0;
        if (i < PRE_NMS) {
            int idx = selIdx[i];
            #pragma unroll
            for (int d = 0; d < 4; ++d)
                bb[i * 4 + d] = (idx >= 0) ? bsrc[(size_t)idx * 4 + d] : 0.f;
        }
    }
    __syncthreads();

    // greedy NMS: sequential over i, parallel over j
    for (int i = 0; i < PRE_NMS; ++i) {
        if (keep[i]) {
            float y1 = bb[i * 4 + 0], x1 = bb[i * 4 + 1];
            float y2 = bb[i * 4 + 2], x2b = bb[i * 4 + 3];
            float ai = (y2 - y1) * (x2b - x1);
            for (int j2 = i + 1 + tid; j2 < PRE_NMS; j2 += 256) {
                if (keep[j2]) {
                    float ty = fmaxf(y1, bb[j2 * 4 + 0]);
                    float tx = fmaxf(x1, bb[j2 * 4 + 1]);
                    float by = fminf(y2, bb[j2 * 4 + 2]);
                    float bx = fminf(x2b, bb[j2 * 4 + 3]);
                    float inter = fmaxf(by - ty, 0.f) * fmaxf(bx - tx, 0.f);
                    float aj = (bb[j2 * 4 + 2] - bb[j2 * 4 + 0]) *
                               (bb[j2 * 4 + 3] - bb[j2 * 4 + 1]);
                    float iou = inter / (ai + aj - inter + 1e-9f);
                    if (iou > 0.7f) keep[j2] = 0;
                }
            }
        }
        __syncthreads();
    }

    // emit first 300 kept boxes (descending score order), zero-pad rest
    if (tid == 0) {
        int r = 0;
        for (int i = 0; i < PRE_NMS && r < POST_NMS; ++i) {
            if (keep[i]) {
                #pragma unroll
                for (int d = 0; d < 4; ++d)
                    rois[((size_t)img * POST_NMS + r) * 4 + d] = bb[i * 4 + d];
                ++r;
            }
        }
        for (; r < POST_NMS; ++r)
            #pragma unroll
            for (int d = 0; d < 4; ++d)
                rois[((size_t)img * POST_NMS + r) * 4 + d] = 0.f;
    }
    for (int r = tid; r < POST_NMS; r += 256)
        roi_idx[img * POST_NMS + r] = img;
}

// ---------------------------------------------------------------------------
extern "C" void kernel_launch(void* const* d_in, const int* in_sizes, int n_in,
                              void* d_out, int out_size, void* d_ws, size_t ws_size,
                              hipStream_t stream)
{
    const float* x       = (const float*)d_in[0];
    const float* conv1_w = (const float*)d_in[1];
    const float* conv1_b = (const float*)d_in[2];
    const float* score_w = (const float*)d_in[3];
    const float* score_b = (const float*)d_in[4];
    const float* loc_w   = (const float*)d_in[5];
    const float* loc_b   = (const float*)d_in[6];

    // d_out flat layout (element offsets, f32 slots):
    float* out        = (float*)d_out;
    float* out_locs   = out;                   // 8*36864*4 = 1179648
    float* out_scores = out + 1179648;         // 8*36864*2 =  589824
    float* out_rois   = out + 1769472;         // 2400*4    =    9600
    int*   out_ridx   = (int*)(out + 1779072); // 2400 (int32 bits)
    float* out_anc    = out + 1781472;         // 36864*4   =  147456

    // workspace layout (bytes)
    char* ws = (char*)d_ws;
    float*    h     = (float*)ws;                              // 67108864 B
    float*    boxes = (float*)(ws + 67108864);                 //  4718592 B
    unsigned* keys  = (unsigned*)(ws + 71827456);              //  1179648 B
    float*    fg    = (float*)(ws + 73007104);                 //  1179648 B
    unsigned* Wp    = (unsigned*)(ws + 74186752);              //  4718592 B
    unsigned* Xp    = (unsigned*)(ws + 78905344);              // 33554432 B

    wpack_k<<<(CIN * 9 * 256 + 255) / 256, 256, 0, stream>>>(conv1_w, Wp);
    xpack_k<<<NIMG * HH, 256, 0, stream>>>(x, Xp);

    dim3 g1(NIMG * HH, CIN / 256);
    conv3x3_wmma<<<g1, 256, 0, stream>>>(Wp, Xp, conv1_b, h);
    conv1x1_head<<<NIMG * HH, 64, 0, stream>>>(h, score_w, score_b, loc_w, loc_b,
                                               out_locs, out_scores, fg);
    anchors_k<<<(ANCN + 255) / 256, 256, 0, stream>>>(out_anc);
    decode_k<<<(NIMG * ANCN + 255) / 256, 256, 0, stream>>>(out_anc, out_locs, fg,
                                                            boxes, keys);
    propose_k<<<NIMG, 256, 0, stream>>>(keys, boxes, out_rois, out_ridx);
}
